// MultiHeadAttention_22204980920937
// MI455X (gfx1250) — matrix operand order repaired and checked by reading, not compiled, not run on hardware
//
#include <hip/hip_runtime.h>

// ---------------- Problem constants ----------------
#define BQ   4
#define SEQ  2048
#define DM   1024
#define NH   16
#define DK   64
#define MTOT (BQ * SEQ)   // 8192

typedef __attribute__((ext_vector_type(16))) __bf16 v16bf;
typedef __attribute__((ext_vector_type(8)))  __bf16 v8bf;
typedef __attribute__((ext_vector_type(8)))  float  v8f;

// ---------------- CDNA5 async global->LDS path (guarded) ----------------
#if defined(__gfx1250__) && __has_builtin(__builtin_amdgcn_global_load_async_to_lds_b128)
#define HAVE_ASYNC_LDS 1
#else
#define HAVE_ASYNC_LDS 0
#endif

#if defined(__gfx1250__) && __has_builtin(__builtin_amdgcn_s_wait_asynccnt)
#define WAIT_ASYNC(n) __builtin_amdgcn_s_wait_asynccnt(n)
#elif defined(__gfx1250__)
#define WAIT_ASYNC(n) asm volatile("s_wait_asynccnt %0" ::"n"(n))
#else
#define WAIT_ASYNC(n)
#endif

// Copy one 16B chunk (8 bf16) global -> LDS.
static __device__ inline void async_cp16(__bf16* lds, const __bf16* g) {
#if HAVE_ASYNC_LDS
    // builtin signature (from compiler diagnostic): param 1 is
    // 'int __vector(4) addrspace(1)*' -- cast both pointers to int4 in the
    // proper address spaces.
    typedef int v4i __attribute__((vector_size(16)));
    typedef __attribute__((address_space(1))) v4i* gptr_t;
    typedef __attribute__((address_space(3))) v4i* lptr_t;
    __builtin_amdgcn_global_load_async_to_lds_b128(
        (gptr_t)(void*)g, (lptr_t)(void*)lds, /*offset=*/0, /*cpol=*/0);
#else
    *(v8bf*)lds = *(const v8bf*)g;   // sync fallback: load + ds_store
#endif
}

static __device__ inline v8f wmma_bf16(v16bf a, v16bf b, v8f c) {
    return __builtin_amdgcn_wmma_f32_16x16x32_bf16(
        /*neg_a=*/false, a, /*neg_b=*/false, b,
        /*c_mod=*/(short)0, c, /*reuse_a=*/false, /*reuse_b=*/false);
}

// A-fragment (16x32 bf16): two 8-element K-chunks per lane.
static __device__ inline v16bf ld_afrag_bf16(const __bf16* p0, const __bf16* p1) {
    union { v16bf v; v8bf h[2]; } u;
    u.h[0] = *(const v8bf*)p0;
    u.h[1] = *(const v8bf*)p1;
    return u.v;
}

// B-fragment (32x16 bf16): two 8-element K-chunks per lane (p, p + 16).
static __device__ inline v16bf ld_bfrag(const __bf16* p) {
    union { v16bf v; v8bf h[2]; } u;
    u.h[0] = *(const v8bf*)p;
    u.h[1] = *(const v8bf*)(p + 16);
    return u.v;
}

static __device__ inline v8bf cvt8(const float* p) {
    v8f f = *(const v8f*)p;       // 2x global_load_b128
    v8bf r;
#pragma unroll
    for (int i = 0; i < 8; ++i) r[i] = (__bf16)f[i];
    return r;
}

// Load 16 contiguous fp32, convert -> 16 bf16.
static __device__ inline v16bf cvt16(const float* p) {
    union { v16bf v; v8bf h[2]; } u;
    u.h[0] = cvt8(p);
    u.h[1] = cvt8(p + 8);
    return u.v;
}

static __device__ inline void st16(__bf16* d, v16bf v) {
    union { v16bf v; v8bf h[2]; } u;
    u.v = v;
    *(v8bf*)d       = u.h[0];
    *(v8bf*)(d + 8) = u.h[1];
}

// ---------------------------------------------------------------------------
// LDS-staged GEMM: Y[m][n] = sum_k A[m][k] * W[n][k] + bias[n]
//   AMODE  : 1 = A fp32 (converted once per block while staging), 0 = A bf16
//            (bf16 A is staged with async global->LDS copies)
//   OUTMODE: 0 = bf16 head layout, 1 = bf16 transposed-V, 2 = fp32 row-major
// Block = 256 threads (8 waves). Block tile 128x128, K-step 32; per wave 64x32.
// ---------------------------------------------------------------------------
#define KP 40   // LDS K-stride (bf16): 80B rows, keeps b128 chunks 16B-aligned
template <int AMODE, int OUTMODE>
__global__ __launch_bounds__(256)
void gemm_wmma_kernel(const void* __restrict__ Ap, const float* __restrict__ W,
                      const float* __restrict__ bias, void* __restrict__ Outp) {
    __shared__ __bf16 aL[128 * KP];
    __shared__ __bf16 wL[128 * KP];

    const int tid    = threadIdx.x;
    const int lane   = tid & 31;
    const int wave   = tid >> 5;
    const int lane16 = lane & 15;
    const int hf     = lane >> 4;

    const int Mblk = blockIdx.y * 128, Nblk = blockIdx.x * 128;
    const int Mw   = (wave & 1) * 64, Nw = (wave >> 1) * 32;
    const int ko0 = hf * 8, ko1 = 16 + hf * 8;   // A-fragment chunks
    const int kb  = hf * 8;                      // B-fragment K chunks

    const int srow = tid >> 1;            // staging: row 0..127
    const int soff = (tid & 1) * 16;      // staging: 16-elem half-row

    const float*  Af = (const float*)Ap;
    const __bf16* Ab = (const __bf16*)Ap;

    v8f acc[4][2] = {};

    for (int kt = 0; kt < DM; kt += 32) {
        // prefetch two K-steps ahead into GL2
        if (kt + 64 < DM) {
            if (AMODE == 1)
                __builtin_prefetch(Af + (size_t)(Mblk + srow) * DM + kt + 64 + soff, 0, 1);
            else
                __builtin_prefetch(Ab + (size_t)(Mblk + srow) * DM + kt + 64 + soff, 0, 1);
            __builtin_prefetch(W + (size_t)(Nblk + srow) * DM + kt + 64 + soff, 0, 1);
        }
        // ---- stage A tile (128x32) ----
        if (AMODE == 1) {
            st16(&aL[srow * KP + soff], cvt16(Af + (size_t)(Mblk + srow) * DM + kt + soff));
        } else {
            const __bf16* g = Ab + (size_t)(Mblk + srow) * DM + kt + soff;
            async_cp16(&aL[srow * KP + soff],     g);
            async_cp16(&aL[srow * KP + soff + 8], g + 8);
        }
        // ---- stage W tile (128x32), fp32 -> bf16 once per block ----
        st16(&wL[srow * KP + soff], cvt16(W + (size_t)(Nblk + srow) * DM + kt + soff));

        if (AMODE == 0) { WAIT_ASYNC(0); }
        __syncthreads();

        v16bf a[4];
#pragma unroll
        for (int mi = 0; mi < 4; ++mi) {
            const __bf16* ar = &aL[(Mw + mi * 16 + lane16) * KP];
            a[mi] = ld_afrag_bf16(ar + ko0, ar + ko1);  // ds_load_b128 x2
        }
        v16bf bfrag[2];
#pragma unroll
        for (int ni = 0; ni < 2; ++ni)
            bfrag[ni] = ld_bfrag(&wL[(Nw + ni * 16 + lane16) * KP + kb]);

#pragma unroll
        for (int mi = 0; mi < 4; ++mi)
#pragma unroll
            for (int ni = 0; ni < 2; ++ni)
                acc[mi][ni] = wmma_bf16(a[mi], bfrag[ni], acc[mi][ni]);

        __syncthreads();
    }

    // Epilogue: C/D layout -> row = r + 8*(lane/16), col = lane16.
#pragma unroll
    for (int mi = 0; mi < 4; ++mi) {
#pragma unroll
        for (int ni = 0; ni < 2; ++ni) {
            const int   n  = Nblk + Nw + ni * 16 + lane16;
            const float bv = bias[n];
#pragma unroll
            for (int r = 0; r < 8; ++r) {
                const int   m = Mblk + Mw + mi * 16 + r + 8 * hf;
                const float v = acc[mi][ni][r] + bv;
                if (OUTMODE == 2) {
                    ((float*)Outp)[(size_t)m * DM + n] = v;
                } else {
                    const int b = m >> 11, s = m & (SEQ - 1);
                    const int h = n >> 6,  dn = n & (DK - 1);
                    const int idx = (OUTMODE == 0)
                        ? (((b * NH + h) * SEQ + s) * DK + dn)
                        : (((b * NH + h) * DK + dn) * SEQ + s);
                    ((__bf16*)Outp)[idx] = (__bf16)v;
                }
            }
        }
    }
}

// ---------------------------------------------------------------------------
// Flash attention. One block owns one (b,h); its 8 waves handle 8 consecutive
// 16-query tiles so K/V tiles are shared. K (32x64) and V^T (64x32) tiles are
// async-copied into double-buffered LDS; next-stage copies overlap compute.
//   qh/kh : (B,H,S,DK) bf16    vt : (B,H,DK,S) bf16    xo : (B,S,DM) bf16
// ---------------------------------------------------------------------------
#define KSTR 72   // K-tile LDS row stride (144B rows, 16B-aligned chunks)
#define VSTR 40   // V-tile LDS row stride (80B rows)
#define PSTR 40   // P-tile LDS row stride
__global__ __launch_bounds__(256)
void attn_wmma_kernel(const __bf16* __restrict__ qh, const __bf16* __restrict__ kh,
                      const __bf16* __restrict__ vt, const int* __restrict__ mask,
                      __bf16* __restrict__ xo) {
    __shared__ __bf16 kL[2][32 * KSTR];
    __shared__ __bf16 vL[2][64 * VSTR];
    __shared__ __bf16 pbuf[8 * 16 * PSTR];

    const int tid    = threadIdx.x;
    const int lane   = tid & 31;
    const int wave   = tid >> 5;
    const int lane16 = lane & 15;
    const int hf     = lane >> 4;

    const int bh = blockIdx.x >> 4;             // one (b,h) per block
    const int qt = (blockIdx.x & 15) * 8 + wave;
    const int b  = bh >> 4, h = bh & 15;

    __bf16* pw = &pbuf[wave * 16 * PSTR];
    const int ko0 = hf * 8, ko1 = 16 + hf * 8, kb = hf * 8;
    const int q0 = qt * 16;

    const __bf16* kbase = kh + (size_t)bh * SEQ * DK;
    const __bf16* vbase = vt + (size_t)bh * DK * SEQ;
    const int*    mrow  = mask + b * SEQ;

    // staging assignment: 256 x 16B K-chunks + 256 x 16B V-chunks, 1+1 per thread
    const int krow = tid >> 3, koff = (tid & 7) * 8;   // K tile: 32 rows x 64 bf16
    const int vrow = tid >> 2, voff = (tid & 3) * 8;   // V tile: 64 rows x 32 bf16

    // Q fragments (registers for the whole pass)
    const __bf16* qrow = qh + ((size_t)bh * SEQ + q0 + lane16) * DK;
    const v16bf a0 = ld_afrag_bf16(qrow + ko0,      qrow + ko1);
    const v16bf a1 = ld_afrag_bf16(qrow + 32 + ko0, qrow + 32 + ko1);

    float mr[8], lr[8];
#pragma unroll
    for (int r = 0; r < 8; ++r) { mr[r] = -1e30f; lr[r] = 0.0f; }
    v8f acc[4] = {};

    // stage tile kt=0 into buffer 0
    async_cp16(&kL[0][krow * KSTR + koff], kbase + (size_t)krow * DK + koff);
    async_cp16(&vL[0][vrow * VSTR + voff], vbase + (size_t)vrow * SEQ + voff);

    for (int kt = 0; kt < SEQ; kt += 32) {
        const int buf = (kt >> 5) & 1;
        if (kt + 64 < SEQ)   // prefetch two tiles ahead into GL2
            __builtin_prefetch(kbase + (size_t)(kt + 64 + krow) * DK + koff, 0, 1);
        if (kt + 32 < SEQ) { // kick off next tile's async copies, then wait for ours
            async_cp16(&kL[buf ^ 1][krow * KSTR + koff],
                       kbase + (size_t)(kt + 32 + krow) * DK + koff);
            async_cp16(&vL[buf ^ 1][vrow * VSTR + voff],
                       vbase + (size_t)vrow * SEQ + kt + 32 + voff);
            WAIT_ASYNC(2);
        } else {
            WAIT_ASYNC(0);
        }
        __syncthreads();

        const __bf16* kl = kL[buf];
        const __bf16* vl = vL[buf];

        // scores: two 16-key tiles, dk=64 reduced over two WMMAs each
        v8f c0 = {}, c1 = {};
        c0 = wmma_bf16(a0, ld_bfrag(kl + lane16 * KSTR + kb), c0);
        c0 = wmma_bf16(a1, ld_bfrag(kl + lane16 * KSTR + 32 + kb), c0);
        c1 = wmma_bf16(a0, ld_bfrag(kl + (16 + lane16) * KSTR + kb), c1);
        c1 = wmma_bf16(a1, ld_bfrag(kl + (16 + lane16) * KSTR + 32 + kb), c1);

        const int mk0 = mrow[kt + lane16];
        const int mk1 = mrow[kt + 16 + lane16];

        float sc[8];
#pragma unroll
        for (int r = 0; r < 8; ++r) {
            float f0 = mk0 ? c0[r] * 0.125f : -1e9f;   // 1/sqrt(64)
            float f1 = mk1 ? c1[r] * 0.125f : -1e9f;
            float t = fmaxf(f0, f1);                   // row max over 16 lanes
            t = fmaxf(t, __shfl_xor(t, 1, 32));
            t = fmaxf(t, __shfl_xor(t, 2, 32));
            t = fmaxf(t, __shfl_xor(t, 4, 32));
            t = fmaxf(t, __shfl_xor(t, 8, 32));
            const float mnew = fmaxf(mr[r], t);
            const float s    = __expf(mr[r] - mnew);
            mr[r] = mnew; sc[r] = s;
            const float p0 = __expf(f0 - mnew);
            const float p1 = __expf(f1 - mnew);
            float rs = p0 + p1;                        // row sum over 16 lanes
            rs += __shfl_xor(rs, 1, 32);
            rs += __shfl_xor(rs, 2, 32);
            rs += __shfl_xor(rs, 4, 32);
            rs += __shfl_xor(rs, 8, 32);
            lr[r] = lr[r] * s + rs;
            const int row = r + 8 * hf;                // stage P for transpose
            pw[row * PSTR + lane16]      = (__bf16)p0;
            pw[row * PSTR + 16 + lane16] = (__bf16)p1;
            acc[0][r] *= s; acc[1][r] *= s; acc[2][r] *= s; acc[3][r] *= s;
        }

        // P as A-fragment (same-wave LDS ops are ordered)
        const v16bf pa = ld_afrag_bf16(pw + lane16 * PSTR + ko0,
                                       pw + lane16 * PSTR + ko1);
#pragma unroll
        for (int nt = 0; nt < 4; ++nt)
            acc[nt] = wmma_bf16(pa, ld_bfrag(vl + (nt * 16 + lane16) * VSTR + kb),
                                acc[nt]);

        __syncthreads();   // all waves done with buf before it is overwritten
    }

    // normalize, write context in (b, s, h*DK + dn) layout
#pragma unroll
    for (int r = 0; r < 8; ++r) {
        const float inv  = 1.0f / lr[r];
        const int   srow = q0 + r + 8 * hf;
#pragma unroll
        for (int nt = 0; nt < 4; ++nt) {
            const int col = h * DK + nt * 16 + lane16;
            xo[((size_t)b * SEQ + srow) * DM + col] = (__bf16)(acc[nt][r] * inv);
        }
    }
}

// ---------------------------------------------------------------------------
extern "C" void kernel_launch(void* const* d_in, const int* in_sizes, int n_in,
                              void* d_out, int out_size, void* d_ws, size_t ws_size,
                              hipStream_t stream) {
    const float* q    = (const float*)d_in[0];
    const float* k    = (const float*)d_in[1];
    const float* v    = (const float*)d_in[2];
    const int*   mask = (const int*)  d_in[3];
    const float* Wq   = (const float*)d_in[4];
    const float* bq   = (const float*)d_in[5];
    const float* Wk   = (const float*)d_in[6];
    const float* bk   = (const float*)d_in[7];
    const float* Wv   = (const float*)d_in[8];
    const float* bv   = (const float*)d_in[9];
    const float* Wo   = (const float*)d_in[10];
    const float* bo   = (const float*)d_in[11];

    const size_t elems = (size_t)MTOT * DM;       // 8M bf16 elems = 16 MB each
    __bf16* qh  = (__bf16*)d_ws;
    __bf16* khd = qh  + elems;
    __bf16* vtd = khd + elems;
    __bf16* xat = vtd + elems;                    // total scratch: 64 MB

    dim3 grid(DM / 128, MTOT / 128), blk(256);
    // Q, K projections -> (B,H,S,DK); V projection -> transposed (B,H,DK,S)
    gemm_wmma_kernel<1, 0><<<grid, blk, 0, stream>>>(q, Wq, bq, qh);
    gemm_wmma_kernel<1, 0><<<grid, blk, 0, stream>>>(k, Wk, bk, khd);
    gemm_wmma_kernel<1, 1><<<grid, blk, 0, stream>>>(v, Wv, bv, vtd);

    // flash attention: one block per (b,h, 8-query-tile slab): 64*16 blocks
    attn_wmma_kernel<<<BQ * NH * (SEQ / 128), blk, 0, stream>>>(
        qh, khd, vtd, mask, xat);

    // output projection (bf16 A staged via async copies) -> fp32 d_out
    gemm_wmma_kernel<0, 2><<<grid, blk, 0, stream>>>(xat, Wo, bo, d_out);
}
